// FDGNN_12902081757490
// MI455X (gfx1250) — compile-verified, matrix operand order checked by reading, not run on hardware
//
#include <hip/hip_runtime.h>
#include <hip/hip_bf16.h>

typedef __attribute__((ext_vector_type(16))) __bf16 v16bf;
typedef __attribute__((ext_vector_type(8)))  __bf16 v8bf;
typedef __attribute__((ext_vector_type(8)))  float  v8f;

#define FD 64  // feature dim

// Hardware tanh (v_tanh_f32 is a native TRANS op on gfx1250); guarded so the
// code still compiles if this toolchain names the builtin differently.
__device__ inline float fast_tanh(float x) {
#if __has_builtin(__builtin_amdgcn_tanhf)
  return __builtin_amdgcn_tanhf(x);
#elif __has_builtin(__builtin_amdgcn_tanh_f32)
  return __builtin_amdgcn_tanh_f32(x);
#else
  return tanhf(x);
#endif
}

// ---------------------------------------------------------------------------
// WMMA fragment layouts (wave32, v_wmma_f32_16x16x32_bf16, ISA 7.12.2):
//   A (16x32 MxK): lane l<16 -> row M=l, K pairs {0..7},{16..23};
//                  lane l>=16 -> row M=l-16, K pairs {8..15},{24..31}
//   B (32x16 KxN): mirrored; lane holds one column N, same K packing.
//   C/D (16x16 f32): vgpr r, lanes 0-15: M=r,N=lane; lanes 16-31: M=8+r,N=l-16
// ---------------------------------------------------------------------------

__device__ inline v16bf load_afrag_bf16(const __bf16* __restrict__ row,
                                        int kbase, int half) {
  v8bf c0 = *(const v8bf*)(row + kbase + half * 8);        // 16B aligned
  v8bf c1 = *(const v8bf*)(row + kbase + 16 + half * 8);
  v16bf a;
#pragma unroll
  for (int i = 0; i < 8; ++i) { a[i] = c0[i]; a[8 + i] = c1[i]; }
  return a;
}

__device__ inline v16bf load_afrag_f32(const float* __restrict__ row,
                                       int kbase, int half) {
  float4 q0 = *(const float4*)(row + kbase + half * 8);
  float4 q1 = *(const float4*)(row + kbase + half * 8 + 4);
  float4 q2 = *(const float4*)(row + kbase + 16 + half * 8);
  float4 q3 = *(const float4*)(row + kbase + 16 + half * 8 + 4);
  v16bf a;
  a[0] = (__bf16)q0.x;  a[1] = (__bf16)q0.y;  a[2] = (__bf16)q0.z;  a[3] = (__bf16)q0.w;
  a[4] = (__bf16)q1.x;  a[5] = (__bf16)q1.y;  a[6] = (__bf16)q1.z;  a[7] = (__bf16)q1.w;
  a[8] = (__bf16)q2.x;  a[9] = (__bf16)q2.y;  a[10] = (__bf16)q2.z; a[11] = (__bf16)q2.w;
  a[12] = (__bf16)q3.x; a[13] = (__bf16)q3.y; a[14] = (__bf16)q3.z; a[15] = (__bf16)q3.w;
  return a;
}

// Pre-packed weight fragment: wpack[(frag*32+lane)*16 .. +15], 32B per lane.
__device__ inline v16bf load_wfrag(const __bf16* __restrict__ wpack,
                                   int frag, int lane) {
  const v8bf* p = (const v8bf*)(wpack + (((long)frag * 32 + lane) << 4));
  v8bf c0 = p[0], c1 = p[1];
  v16bf b;
#pragma unroll
  for (int i = 0; i < 8; ++i) { b[i] = c0[i]; b[8 + i] = c1[i]; }
  return b;
}

// ---------------------------------------------------------------------------
// Weight fragment packer — run once per launch. 22 fragments:
//   0..3   wm1 [64x32]: frag = kc*2 + t        (kc = K chunk, t = N tile)
//   4..7   wm2 [32x64]: frag = 4 + t
//   8..9   wu1 [64x16]: frag = 8 + kc
//   10..13 wu2 [16x64]: frag = 10 + t  (K rows >= 16 zero-padded)
//   14..21 wo  [64x64]: frag = 14 + kc*4 + t
// ---------------------------------------------------------------------------
#define NFRAG 22

__global__ void __launch_bounds__(256)
fdgnn_pack_weights(const float* __restrict__ wm1, const float* __restrict__ wm2,
                   const float* __restrict__ wu1, const float* __restrict__ wu2,
                   const float* __restrict__ wo, __bf16* __restrict__ wpack) {
  int idx = blockIdx.x * blockDim.x + threadIdx.x;  // frag*32 + lane
  if (idx >= NFRAG * 32) return;
  int frag = idx >> 5, lane = idx & 31;
  int half = lane >> 4, nl = lane & 15;
  const float* W;
  int ldw, kbase, kmax, t;
  if (frag < 4)       { W = wm1; ldw = 32; kmax = 64; kbase = (frag >> 1) * 32; t = frag & 1; }
  else if (frag < 8)  { W = wm2; ldw = 64; kmax = 32; kbase = 0;               t = frag - 4; }
  else if (frag < 10) { W = wu1; ldw = 16; kmax = 64; kbase = (frag - 8) * 32; t = 0; }
  else if (frag < 14) { W = wu2; ldw = 64; kmax = 16; kbase = 0;               t = frag - 10; }
  else { int f = frag - 14; W = wo; ldw = 64; kmax = 64; kbase = (f >> 2) * 32; t = f & 3; }
  int ncol = t * 16 + nl;
  __bf16* out = wpack + ((long)idx << 4);
#pragma unroll
  for (int i = 0; i < 8; ++i) {
    int k0 = kbase + half * 8 + i;
    int k1 = kbase + 16 + half * 8 + i;
    out[i]     = (__bf16)((k0 < kmax) ? W[k0 * ldw + ncol] : 0.0f);
    out[8 + i] = (__bf16)((k1 < kmax) ? W[k1 * ldw + ncol] : 0.0f);
  }
}

// ---------------------------------------------------------------------------
// Fused 2-layer MLP: Y = relu(relu(X@W1+b1)@W2+b2); bf16 out, f32 WMMA accum.
// One wave per 16-row tile, 8 waves/block. In-place (Y==X) safe: a wave loads
// its 16 rows fully before storing them. Hidden + output staged via LDS.
// ---------------------------------------------------------------------------
template <bool IN_F32>
__global__ void __launch_bounds__(256)
fdgnn_mlp2_wmma(const void* __restrict__ Xv, __bf16* __restrict__ Y,
                const __bf16* __restrict__ wpack,
                const float* __restrict__ B1, const float* __restrict__ B2,
                int l1base, int l2base, int H, int nrows) {
  __shared__ float hid[8][16 * 32];  // per-wave; reused as 16x64 bf16 for out
  const int wave = threadIdx.x >> 5;
  const int lane = threadIdx.x & 31;
  const int half = lane >> 4;
  const int nl   = lane & 15;
  const int rowBase = (blockIdx.x * 8 + wave) * 16;
  if (rowBase >= nrows) return;  // wave-uniform: EXEC all-ones for WMMA

  int mrow = rowBase + nl;
  if (mrow >= nrows) mrow = nrows - 1;  // N is a multiple of 16; safety only
  v16bf a_lo, a_hi;
  if (IN_F32) {
    const float* row = (const float*)Xv + (long)mrow * FD;
    a_lo = load_afrag_f32(row, 0, half);
    a_hi = load_afrag_f32(row, 32, half);
  } else {
    const __bf16* row = (const __bf16*)Xv + (long)mrow * FD;
    a_lo = load_afrag_bf16(row, 0, half);
    a_hi = load_afrag_bf16(row, 32, half);
  }

  float* hw = hid[wave];
  const int nt = H >> 4;  // hidden N tiles (2 or 1)

  // ---- layer 1: hidden[16 x H] -> LDS (K range [H,32) zero-padded) ----
#pragma unroll
  for (int t = 0; t < 2; ++t) {
    if (t < nt) {  // wave-uniform
      v16bf b0 = load_wfrag(wpack, l1base + t, lane);       // K 0..31
      v16bf b1 = load_wfrag(wpack, l1base + nt + t, lane);  // K 32..63
      float bias = B1[t * 16 + nl];
      v8f c = {bias, bias, bias, bias, bias, bias, bias, bias};
      c = __builtin_amdgcn_wmma_f32_16x16x32_bf16(false, a_lo, false, b0,
                                                  (short)0, c, false, false);
      c = __builtin_amdgcn_wmma_f32_16x16x32_bf16(false, a_hi, false, b1,
                                                  (short)0, c, false, false);
#pragma unroll
      for (int r = 0; r < 8; ++r) {
        float v = c[r];
        v = v > 0.0f ? v : 0.0f;
        hw[(r + half * 8) * 32 + t * 16 + nl] = v;  // D-layout -> row-major
      }
    } else {
#pragma unroll
      for (int r = 0; r < 8; ++r) hw[(r + half * 8) * 32 + t * 16 + nl] = 0.0f;
    }
  }
  __syncthreads();

  // ---- layer 2 A fragment from LDS hidden ----
  v16bf a2;
  {
    const float* hrow = hw + nl * 32;
#pragma unroll
    for (int i = 0; i < 8; ++i) {
      a2[i]     = (__bf16)hrow[half * 8 + i];
      a2[8 + i] = (__bf16)hrow[16 + half * 8 + i];
    }
  }

  // ---- layer 2: 4 N tiles; stage bf16 output tile in (reused) LDS ----
  __bf16* hb = (__bf16*)hw;  // 16 x 64 bf16 (2KB, same footprint)
#pragma unroll
  for (int t = 0; t < 4; ++t) {
    v16bf b2 = load_wfrag(wpack, l2base + t, lane);
    float bias = B2[t * 16 + nl];
    v8f c = {bias, bias, bias, bias, bias, bias, bias, bias};
    c = __builtin_amdgcn_wmma_f32_16x16x32_bf16(false, a2, false, b2,
                                                (short)0, c, false, false);
#pragma unroll
    for (int r = 0; r < 8; ++r) {
      float v = c[r];
      v = v > 0.0f ? v : 0.0f;
      hb[(r + half * 8) * FD + t * 16 + nl] = (__bf16)v;
    }
  }
  __syncthreads();

  // ---- vectorized row-major writeback: 128 x 16B chunks per wave ----
#pragma unroll
  for (int r = 0; r < 4; ++r) {
    int chunk = lane + 32 * r;        // 0..127
    int row = chunk >> 3;             // 16 rows
    int cir = chunk & 7;              // 8 x 8-bf16 chunks per row
    int grow = rowBase + row;
    if (grow < nrows)
      *(v8bf*)(Y + (long)grow * FD + cir * 8) =
          *(const v8bf*)(hb + row * FD + cir * 8);
  }
}

// ---------------------------------------------------------------------------
// Head: out = tanh(X @ Wo + bo), bf16 in, f32 out (v_tanh_f32 epilogue).
// Output tile staged in LDS for b128 row-major writeback.
// ---------------------------------------------------------------------------
__global__ void __launch_bounds__(256)
fdgnn_head_wmma(const __bf16* __restrict__ X, const __bf16* __restrict__ wpack,
                const float* __restrict__ B, float* __restrict__ Y, int nrows) {
  __shared__ float outb[8][16 * FD];  // 32 KB / block (WGP has 320 KB)
  const int wave = threadIdx.x >> 5;
  const int lane = threadIdx.x & 31;
  const int half = lane >> 4;
  const int nl   = lane & 15;
  const int rowBase = (blockIdx.x * 8 + wave) * 16;
  if (rowBase >= nrows) return;

  int mrow = rowBase + nl;
  if (mrow >= nrows) mrow = nrows - 1;
  const __bf16* row = X + (long)mrow * FD;
  v16bf a_lo = load_afrag_bf16(row, 0, half);
  v16bf a_hi = load_afrag_bf16(row, 32, half);

  float* ob = outb[wave];
#pragma unroll
  for (int t = 0; t < 4; ++t) {
    v16bf b0 = load_wfrag(wpack, 14 + t, lane);      // K 0..31
    v16bf b1 = load_wfrag(wpack, 14 + 4 + t, lane);  // K 32..63
    float bias = B[t * 16 + nl];
    v8f c = {bias, bias, bias, bias, bias, bias, bias, bias};
    c = __builtin_amdgcn_wmma_f32_16x16x32_bf16(false, a_lo, false, b0,
                                                (short)0, c, false, false);
    c = __builtin_amdgcn_wmma_f32_16x16x32_bf16(false, a_hi, false, b1,
                                                (short)0, c, false, false);
#pragma unroll
    for (int r = 0; r < 8; ++r)
      ob[(r + half * 8) * FD + t * 16 + nl] = fast_tanh(c[r]);
  }
  __syncthreads();

  // 256 x 16B chunks per wave: 16 rows x 16 float4 chunks
#pragma unroll
  for (int r = 0; r < 8; ++r) {
    int chunk = lane + 32 * r;  // 0..255
    int row16 = chunk >> 4;
    int cir = chunk & 15;
    int grow = rowBase + row16;
    if (grow < nrows)
      *(float4*)(Y + (long)grow * FD + cir * 4) =
          *(const float4*)(ob + row16 * FD + cir * 4);
  }
}

// ---------------------------------------------------------------------------
// CSR build (edges are static: built once per launch, reused for 3 rounds).
// ---------------------------------------------------------------------------
__global__ void __launch_bounds__(256)
fdgnn_zero_i32(int* __restrict__ p, int n) {
  int i = blockIdx.x * blockDim.x + threadIdx.x;
  if (i < n) p[i] = 0;
}

__global__ void __launch_bounds__(256)
fdgnn_hist(const long long* __restrict__ dst, int* __restrict__ counts, int E) {
  int e = blockIdx.x * blockDim.x + threadIdx.x;
  if (e < E) atomicAdd(&counts[(int)dst[e]], 1);
}

// Single-block chunked Hillis-Steele exclusive scan; also seeds cursor.
__global__ void __launch_bounds__(1024)
fdgnn_scan(const int* __restrict__ counts, int* __restrict__ rowptr,
           int* __restrict__ cursor, int n) {
  __shared__ int lds[1024];
  __shared__ int carry;
  const int t = threadIdx.x;
  if (t == 0) carry = 0;
  __syncthreads();
  for (int base = 0; base < n; base += 1024) {
    int v = (base + t < n) ? counts[base + t] : 0;
    lds[t] = v;
    __syncthreads();
#pragma unroll
    for (int off = 1; off < 1024; off <<= 1) {
      int x = (t >= off) ? lds[t - off] : 0;
      __syncthreads();
      lds[t] += x;
      __syncthreads();
    }
    int excl = lds[t] - v + carry;
    if (base + t < n) { rowptr[base + t] = excl; cursor[base + t] = excl; }
    __syncthreads();
    if (t == 0) carry += lds[1023];
    __syncthreads();
  }
  if (t == 0) rowptr[n] = carry;
}

__global__ void __launch_bounds__(256)
fdgnn_fill(const long long* __restrict__ src, const long long* __restrict__ dst,
           int* __restrict__ cursor, int* __restrict__ col, int E) {
  int e = blockIdx.x * blockDim.x + threadIdx.x;
  if (e < E) {
    int pos = atomicAdd(&cursor[(int)dst[e]], 1);
    col[pos] = (int)src[e];
  }
}

// ---------------------------------------------------------------------------
// Atomic-free aggregation: one wave per dst node; lane owns 2 feature cols.
// msg rows (128B, L2-resident) read 4B/lane; f32 register accumulation.
// 2x unrolled so two row reads are in flight per loop iteration.
// ---------------------------------------------------------------------------
__global__ void __launch_bounds__(256)
fdgnn_gather_sum(const __bf16* __restrict__ msg, const int* __restrict__ rowptr,
                 const int* __restrict__ col, __bf16* __restrict__ out,
                 int ndst) {
  const int wave = threadIdx.x >> 5;
  const int lane = threadIdx.x & 31;
  const int node = blockIdx.x * 8 + wave;
  if (node >= ndst) return;
  const int beg = rowptr[node], end = rowptr[node + 1];
  float ax = 0.0f, ay = 0.0f;
  union Pk { unsigned u; __bf16 h[2]; };
  int j = beg;
  for (; j + 1 < end; j += 2) {
    int s0 = col[j], s1 = col[j + 1];
    Pk p0, p1;
    p0.u = *(const unsigned*)(msg + (long)s0 * FD + lane * 2);
    p1.u = *(const unsigned*)(msg + (long)s1 * FD + lane * 2);
    ax += (float)p0.h[0] + (float)p1.h[0];
    ay += (float)p0.h[1] + (float)p1.h[1];
  }
  if (j < end) {
    Pk p;
    p.u = *(const unsigned*)(msg + (long)col[j] * FD + lane * 2);
    ax += (float)p.h[0];
    ay += (float)p.h[1];
  }
  Pk o;
  o.h[0] = (__bf16)ax;
  o.h[1] = (__bf16)ay;
  *(unsigned*)(out + (long)node * FD + lane * 2) = o.u;
}

// ---------------------------------------------------------------------------

extern "C" void kernel_launch(void* const* d_in, const int* in_sizes, int n_in,
                              void* d_out, int out_size, void* d_ws, size_t ws_size,
                              hipStream_t stream) {
  const float* xs0 = (const float*)d_in[0];
  const float* xi0 = (const float*)d_in[1];
  const long long* e_s2i = (const long long*)d_in[2];  // [2,E]: row0=src, row1=dst
  const long long* e_i2s = (const long long*)d_in[3];
  const float* wm1 = (const float*)d_in[4];
  const float* bm1 = (const float*)d_in[5];
  const float* wm2 = (const float*)d_in[6];
  const float* bm2 = (const float*)d_in[7];
  const float* wu1 = (const float*)d_in[8];
  const float* bu1 = (const float*)d_in[9];
  const float* wu2 = (const float*)d_in[10];
  const float* bu2 = (const float*)d_in[11];
  const float* wo  = (const float*)d_in[12];
  const float* bo  = (const float*)d_in[13];

  const int NS = in_sizes[0] / FD;
  const int NI = in_sizes[1] / FD;
  const int E  = in_sizes[2] / 2;
  const int NMAX = NS > NI ? NS : NI;

  // -------- workspace carve-up (~74 MB total) --------
  char* w = (char*)d_ws;
  auto carve = [&](size_t bytes) {
    char* p = w;
    w += (bytes + 255) & ~(size_t)255;
    return p;
  };
  const size_t capB = (size_t)NMAX * FD * sizeof(__bf16);
  __bf16* P0   = (__bf16*)carve(capB);
  __bf16* P1   = (__bf16*)carve(capB);
  __bf16* P2   = (__bf16*)carve(capB);
  __bf16* P3   = (__bf16*)carve(capB);
  __bf16* MSG  = (__bf16*)carve(capB);
  __bf16* WPK  = (__bf16*)carve((size_t)NFRAG * 32 * 16 * sizeof(__bf16));
  int* counts  = (int*)carve((size_t)NMAX * 4);
  int* rpS     = (int*)carve((size_t)(NMAX + 1) * 4);  // s2i CSR (dst = interfered)
  int* curS    = (int*)carve((size_t)NMAX * 4);
  int* colS    = (int*)carve((size_t)E * 4);
  int* rpI     = (int*)carve((size_t)(NMAX + 1) * 4);  // i2s CSR (dst = served)
  int* curI    = (int*)carve((size_t)NMAX * 4);
  int* colI    = (int*)carve((size_t)E * 4);

  const dim3 blk(256);
  auto g1 = [](long n) { return dim3((unsigned)((n + 255) / 256)); };
  auto mlpGrid = [](int n) { return dim3((unsigned)((n + 127) / 128)); };

  // -------- weight fragment packing --------
  fdgnn_pack_weights<<<g1(NFRAG * 32), blk, 0, stream>>>(wm1, wm2, wu1, wu2, wo, WPK);

  // -------- CSR build (once; reused by all 3 rounds) --------
  auto buildCSR = [&](const long long* edges, int ndst, int* rp, int* cur, int* cl) {
    fdgnn_zero_i32<<<g1(ndst), blk, 0, stream>>>(counts, ndst);
    fdgnn_hist<<<g1(E), blk, 0, stream>>>(edges + E, counts, E);
    fdgnn_scan<<<1, 1024, 0, stream>>>(counts, rp, cur, ndst);
    fdgnn_fill<<<g1(E), blk, 0, stream>>>(edges, edges + E, cur, cl, E);
  };
  buildCSR(e_s2i, NI, rpS, curS, colS);
  buildCSR(e_i2s, NS, rpI, curI, colI);

  // -------- conv: node-level message MLP (hoisted off edges), gather, update
  auto conv = [&](const void* xsrc, bool srcF32, int nsrc, const int* rp,
                  const int* cl, int ndst, __bf16* dstbuf) {
    if (srcF32)
      fdgnn_mlp2_wmma<true><<<mlpGrid(nsrc), blk, 0, stream>>>(
          xsrc, MSG, WPK, bm1, bm2, 0, 4, 32, nsrc);
    else
      fdgnn_mlp2_wmma<false><<<mlpGrid(nsrc), blk, 0, stream>>>(
          xsrc, MSG, WPK, bm1, bm2, 0, 4, 32, nsrc);
    fdgnn_gather_sum<<<dim3((unsigned)((ndst + 7) / 8)), blk, 0, stream>>>(
        MSG, rp, cl, dstbuf, ndst);
    fdgnn_mlp2_wmma<false><<<mlpGrid(ndst), blk, 0, stream>>>(  // in place
        dstbuf, dstbuf, WPK, bu1, bu2, 8, 10, 16, ndst);
  };

  // round 0 (f32 inputs)
  conv(xs0, true, NS, rpS, colS, NI, P0);  // new_i -> P0
  conv(xi0, true, NI, rpI, colI, NS, P1);  // new_s -> P1
  // round 1
  conv(P1, false, NS, rpS, colS, NI, P2);
  conv(P0, false, NI, rpI, colI, NS, P3);
  // round 2
  conv(P3, false, NS, rpS, colS, NI, P0);
  conv(P2, false, NI, rpI, colI, NS, P1);

  // head: tanh(xs @ wo + bo) -> f32 output
  fdgnn_head_wmma<<<mlpGrid(NS), blk, 0, stream>>>(P1, WPK, bo, (float*)d_out, NS);
}
